// BBPSeq2SeqAttn_60979945669283
// MI455X (gfx1250) — matrix operand order repaired and checked by reading, not compiled, not run on hardware
//
#include <hip/hip_runtime.h>

// ---------------------------------------------------------------------------
// BBPSeq2SeqAttn for MI455X (gfx1250), wave32, WMMA bf16 16x16x32.
//
// Roofline: logits GEMM = 4096x32000x256 = 67 GFLOP, 512MB fp32 output
// (~22us store floor at 23.3 TB/s). W_out bf16 (16MB) is L2-resident.
// Recurrent scans are latency-bound single-workgroup loops using WMMA with
// state held in LDS. Logits kernel stages A via the CDNA5 async
// global->LDS DMA (ASYNCcnt) and uses 4x4 register tiles per wave.
// ---------------------------------------------------------------------------

#define DD   256
#define BB   32
#define TSRC 128
#define TTGT 128
#define VDE  32000

typedef __bf16 bf16;
typedef __attribute__((ext_vector_type(16))) __bf16 v16bf;
typedef __attribute__((ext_vector_type(8)))  __bf16 v8bf;
typedef __attribute__((ext_vector_type(8)))  float  v8f;

__device__ __forceinline__ v8f zero8() {
  v8f z;
#pragma unroll
  for (int q = 0; q < 8; ++q) z[q] = 0.0f;
  return z;
}

// A-fragment (16x32 bf16, M x K), row-major source with row stride DD.
// Caller passes base already offset to (m_tile row 0, k_tile col 0).
// Lane m=lane&15, hi=lane>>4 holds K = 16*hi + j (matches B-frag order)
// ->  two contiguous 8-elem (16B) chunks.
__device__ __forceinline__ v16bf a_frag(const bf16* A, int lane) {
  int m = lane & 15, hi = (lane >> 4) & 1;
  const bf16* p = A + m * DD + hi * 16;
  v8bf lo = *(const v8bf*)p;
  v8bf up = *(const v8bf*)(p + 8);
  return __builtin_shufflevector(lo, up, 0, 1, 2, 3, 4, 5, 6, 7,
                                 8, 9, 10, 11, 12, 13, 14, 15);
}

// B-fragment (32x16 bf16, K x N) from a weight stored row-major as W[n][k]
// (i.e. B(k,n) = W[n*DD + k]). Caller passes base already offset to
// (n_tile row 0, k_tile col 0). Order used here (same as a_frag): lane n=lane&15 holds column n,
// element j <-> k = 16*hi + j  ->  one contiguous 32B load per lane.
__device__ __forceinline__ v16bf b_frag(const bf16* W, int lane) {
  int n = lane & 15, hi = (lane >> 4) & 1;
  return *(const v16bf*)(W + (size_t)n * DD + hi * 16);
}

__device__ __forceinline__ v8f wmma_bf16(v16bf a, v16bf b, v8f c) {
  return __builtin_amdgcn_wmma_f32_16x16x32_bf16(false, a, false, b,
                                                 (short)0, c, false, false);
}

// ---------------------------------------------------------------------------
// prep: fp32 -> bf16 weight conversion.
//   Ut[n*256+k]  = U[k*256+n]        (transposed so B-frag loads are contig)
//   Wc[n*256+k]  = ctx_W[n*256+k]    (out = ctx @ ctx_W^T -> B(k,n)=ctx_W[n,k])
//   Wo[v*256+k]  = W_out[v*256+k]    (logits = outs @ W_out^T)
// ---------------------------------------------------------------------------
__global__ __launch_bounds__(256) void prep_kernel(
    const float* __restrict__ U, const float* __restrict__ Wcf,
    const float* __restrict__ Wof, bf16* __restrict__ Ut,
    bf16* __restrict__ Wc, bf16* __restrict__ Wo) {
  size_t stride = (size_t)gridDim.x * blockDim.x;
  for (size_t i = (size_t)blockIdx.x * blockDim.x + threadIdx.x;
       i < (size_t)VDE * DD; i += stride) {
    Wo[i] = (bf16)Wof[i];
    if (i < (size_t)DD * DD) {
      Wc[i] = (bf16)Wcf[i];
      int k = (int)(i >> 8), n = (int)(i & 255);
      Ut[n * DD + k] = (bf16)U[i];
    }
  }
}

// ---------------------------------------------------------------------------
// encoder: 2-layer tanh RNN scan, single workgroup of 8 waves.
// h1,h2 (32x256) live in LDS as bf16. Per GEMM: 2 m-tiles x 16 n-tiles
// = 32 WMMA tiles, 4 per wave, K=256 in 8 wmma steps. H[t] = h2 (fp32).
// ---------------------------------------------------------------------------
__global__ __launch_bounds__(256) void encoder_kernel(
    const bf16* __restrict__ Ut, const float* __restrict__ Een,
    const float* __restrict__ b1, const float* __restrict__ b2,
    const int* __restrict__ src, float* __restrict__ H) {
  __shared__ __align__(16) bf16 h1[BB * DD];
  __shared__ __align__(16) bf16 h2[BB * DD];
  const int tid = threadIdx.x, lane = tid & 31, w = tid >> 5;
  for (int i = tid; i < BB * DD; i += 256) {
    h1[i] = (bf16)0.0f;
    h2[i] = (bf16)0.0f;
  }
  __syncthreads();
  const int mt = (w >> 2) * 16;      // waves 0-3 -> rows 0..15, 4-7 -> 16..31
  const int ntb = (w & 3) * 64;      // 4 n-tiles per wave
  const int hi = lane >> 4, nl = lane & 15;

  for (int t = 0; t < TSRC; ++t) {
    // ---- layer 1: h1 = tanh(h1 @ U + x + b1) ----
    v8f acc[4];
#pragma unroll
    for (int i = 0; i < 4; ++i) acc[i] = zero8();
    for (int k = 0; k < DD; k += 32) {
      v16bf a = a_frag(h1 + mt * DD + k, lane);
#pragma unroll
      for (int i = 0; i < 4; ++i) {
        v16bf b = b_frag(Ut + (size_t)(ntb + i * 16) * DD + k, lane);
        acc[i] = wmma_bf16(a, b, acc[i]);
      }
    }
    float res[4][8];
#pragma unroll
    for (int i = 0; i < 4; ++i) {
#pragma unroll
      for (int r = 0; r < 8; ++r) {
        int row = mt + hi * 8 + r;   // batch
        int col = ntb + i * 16 + nl; // feature
        int tok = src[t * BB + row];
        float x = Een[(size_t)tok * DD + col];
        res[i][r] = tanhf(acc[i][r] + x + b1[col]);
      }
    }
    __syncthreads();
#pragma unroll
    for (int i = 0; i < 4; ++i)
#pragma unroll
      for (int r = 0; r < 8; ++r)
        h1[(mt + hi * 8 + r) * DD + ntb + i * 16 + nl] = (bf16)res[i][r];
    __syncthreads();

    // ---- layer 2: h2 = tanh(h2 @ U + h1_new + b2) ----
#pragma unroll
    for (int i = 0; i < 4; ++i) acc[i] = zero8();
    for (int k = 0; k < DD; k += 32) {
      v16bf a = a_frag(h2 + mt * DD + k, lane);
#pragma unroll
      for (int i = 0; i < 4; ++i) {
        v16bf b = b_frag(Ut + (size_t)(ntb + i * 16) * DD + k, lane);
        acc[i] = wmma_bf16(a, b, acc[i]);
      }
    }
#pragma unroll
    for (int i = 0; i < 4; ++i) {
#pragma unroll
      for (int r = 0; r < 8; ++r) {
        int row = mt + hi * 8 + r;
        int col = ntb + i * 16 + nl;
        res[i][r] = tanhf(acc[i][r] + (float)h1[row * DD + col] + b2[col]);
      }
    }
    __syncthreads();
#pragma unroll
    for (int i = 0; i < 4; ++i)
#pragma unroll
      for (int r = 0; r < 8; ++r) {
        int row = mt + hi * 8 + r;
        int col = ntb + i * 16 + nl;
        h2[row * DD + col] = (bf16)res[i][r];
        H[(size_t)t * BB * DD + row * DD + col] = res[i][r];
      }
    __syncthreads();
  }
}

// ---------------------------------------------------------------------------
// decoder: attention RNN scan, single workgroup of 8 waves.
// outs written as bf16 [t*B+b][d] for the logits GEMM A-matrix.
// ---------------------------------------------------------------------------
__global__ __launch_bounds__(256) void decoder_kernel(
    const bf16* __restrict__ Ut, const bf16* __restrict__ Wc,
    const float* __restrict__ Ede, const float* __restrict__ bdec,
    const int* __restrict__ src, const int* __restrict__ tgt,
    const float* __restrict__ H, bf16* __restrict__ outs) {
  __shared__ __align__(16) bf16 hb[BB * DD];     // 16 KB
  __shared__ __align__(16) bf16 cx[BB * DD];     // 16 KB
  __shared__ float alpha[BB * TSRC];             // 16 KB
  const int tid = threadIdx.x, lane = tid & 31, w = tid >> 5;
  // hT = H[T_src-1]
  for (int i = tid; i < BB * DD; i += 256)
    hb[i] = (bf16)H[(size_t)(TSRC - 1) * BB * DD + i];
  __syncthreads();
  const int mt = (w >> 2) * 16;
  const int ntb = (w & 3) * 64;
  const int hi = lane >> 4, nl = lane & 15;

  for (int t = 0; t < TTGT; ++t) {
    // ---- h = tanh(h @ U + x + b_dec) ----
    v8f acc[4];
#pragma unroll
    for (int i = 0; i < 4; ++i) acc[i] = zero8();
    for (int k = 0; k < DD; k += 32) {
      v16bf a = a_frag(hb + mt * DD + k, lane);
#pragma unroll
      for (int i = 0; i < 4; ++i) {
        v16bf b = b_frag(Ut + (size_t)(ntb + i * 16) * DD + k, lane);
        acc[i] = wmma_bf16(a, b, acc[i]);
      }
    }
    float res[4][8];
#pragma unroll
    for (int i = 0; i < 4; ++i) {
#pragma unroll
      for (int r = 0; r < 8; ++r) {
        int row = mt + hi * 8 + r;
        int col = ntb + i * 16 + nl;
        int tok = tgt[t * BB + row];
        res[i][r] = tanhf(acc[i][r] + Ede[(size_t)tok * DD + col] + bdec[col]);
      }
    }
    __syncthreads();
#pragma unroll
    for (int i = 0; i < 4; ++i)
#pragma unroll
      for (int r = 0; r < 8; ++r)
        hb[(mt + hi * 8 + r) * DD + ntb + i * 16 + nl] = (bf16)res[i][r];
    __syncthreads();

    // ---- scores[b][ts] = (H[ts,b,:] . h[b,:]) / 16, mask PAD ----
    for (int s = 0; s < 16; ++s) {
      int idx = tid * 16 + s;        // 0..4095
      int b = idx >> 7;
      int ts = idx & 127;
      const float* Hp = H + (size_t)ts * BB * DD + b * DD;
      const bf16* hp = hb + b * DD;
      float dot = 0.0f;
      for (int d0 = 0; d0 < DD; d0 += 4) {
        dot += Hp[d0 + 0] * (float)hp[d0 + 0];
        dot += Hp[d0 + 1] * (float)hp[d0 + 1];
        dot += Hp[d0 + 2] * (float)hp[d0 + 2];
        dot += Hp[d0 + 3] * (float)hp[d0 + 3];
      }
      dot *= 0.0625f;
      if (src[ts * BB + b] == 0) dot = -1e9f;
      alpha[b * TSRC + ts] = dot;
    }
    __syncthreads();

    // ---- softmax over T_src, one thread per batch row ----
    if (tid < BB) {
      float mx = -3.4e38f;
      for (int ts = 0; ts < TSRC; ++ts)
        mx = fmaxf(mx, alpha[tid * TSRC + ts]);
      float sum = 0.0f;
      for (int ts = 0; ts < TSRC; ++ts) {
        float e = __expf(alpha[tid * TSRC + ts] - mx);
        alpha[tid * TSRC + ts] = e;
        sum += e;
      }
      float inv = 1.0f / sum;
      for (int ts = 0; ts < TSRC; ++ts) alpha[tid * TSRC + ts] *= inv;
    }
    __syncthreads();

    // ---- ctx[b][d] = sum_ts alpha[b][ts] * H[ts][b][d] ----
    {
      int b = tid >> 3;
      int d0 = (tid & 7) * 32;
      float a2[32];
#pragma unroll
      for (int q = 0; q < 32; ++q) a2[q] = 0.0f;
      for (int ts = 0; ts < TSRC; ++ts) {
        float al = alpha[b * TSRC + ts];
        const float4* Hp4 =
            (const float4*)(H + (size_t)ts * BB * DD + b * DD + d0);
#pragma unroll
        for (int q = 0; q < 8; ++q) {
          float4 v = Hp4[q];
          a2[4 * q + 0] += al * v.x;
          a2[4 * q + 1] += al * v.y;
          a2[4 * q + 2] += al * v.z;
          a2[4 * q + 3] += al * v.w;
        }
      }
#pragma unroll
      for (int q = 0; q < 32; ++q) cx[b * DD + d0 + q] = (bf16)a2[q];
    }
    __syncthreads();

    // ---- out = h + ctx @ ctx_W^T  -> outs (bf16) ----
#pragma unroll
    for (int i = 0; i < 4; ++i) acc[i] = zero8();
    for (int k = 0; k < DD; k += 32) {
      v16bf a = a_frag(cx + mt * DD + k, lane);
#pragma unroll
      for (int i = 0; i < 4; ++i) {
        v16bf b = b_frag(Wc + (size_t)(ntb + i * 16) * DD + k, lane);
        acc[i] = wmma_bf16(a, b, acc[i]);
      }
    }
#pragma unroll
    for (int i = 0; i < 4; ++i) {
#pragma unroll
      for (int r = 0; r < 8; ++r) {
        int row = mt + hi * 8 + r;
        int col = ntb + i * 16 + nl;
        float v = acc[i][r] + (float)hb[row * DD + col];
        outs[(size_t)(t * BB + row) * DD + col] = (bf16)v;
      }
    }
    __syncthreads();
  }
}

// ---------------------------------------------------------------------------
// logits: outs(4096x256 bf16) @ Wo^T -> (4096x32000 fp32).
// Block = 64(M) x 512(N); 8 waves, each owning a 64(M)x64(N) register tile
// (16 v8f accumulators). A staged in LDS via CDNA5 async global->LDS DMA
// (GLOBAL_LOAD_ASYNC_TO_LDS_B128, ASYNCcnt); B streamed from L2-resident
// bf16 W_out. Per K-step: 4 A frags + 4 B frags -> 16 WMMAs.
// Grid.x = 63 covers N=32000 with an edge guard (62.5 full blocks).
// ---------------------------------------------------------------------------
__global__ __launch_bounds__(256) void logits_kernel(
    const bf16* __restrict__ outs, const bf16* __restrict__ Wo,
    float* __restrict__ out) {
  __shared__ __align__(16) bf16 At[64 * DD];  // 32 KB
  const int tid = threadIdx.x, lane = tid & 31, w = tid >> 5;
  const int m0 = blockIdx.y * 64;
  const int n0 = blockIdx.x * 512;
  // ---- stage A block (64 x 256 bf16 = 2048 x 16B) via async DMA to LDS ----
  {
    unsigned long long gbase =
        (unsigned long long)(size_t)(outs + (size_t)m0 * DD);
    unsigned lbase = (unsigned)(size_t)(At);  // low 32 bits = LDS offset
#pragma unroll
    for (int i = 0; i < 8; ++i) {
      int e = tid + i * 256;                  // 16B chunk index, 0..2047
      unsigned lofs = lbase + (unsigned)e * 16u;
      unsigned long long ga = gbase + (unsigned long long)e * 16ull;
      asm volatile("global_load_async_to_lds_b128 %0, %1, off"
                   :: "v"(lofs), "v"(ga)
                   : "memory");
    }
    asm volatile("s_wait_asynccnt 0x0" ::: "memory");
  }
  __syncthreads();

  const int ntb = n0 + w * 64;               // this wave's 4 n-tiles
  const int hi = lane >> 4, nl = lane & 15;
  v8f acc[4][4];
#pragma unroll
  for (int mi = 0; mi < 4; ++mi)
#pragma unroll
    for (int i = 0; i < 4; ++i) acc[mi][i] = zero8();

  for (int k = 0; k < DD; k += 32) {
    // prefetch next K-slice of this wave's B rows (global_prefetch path)
    if (k + 32 < DD) {
      int pn = ntb + nl * 4;
      pn = pn < VDE ? pn : VDE - 1;
      __builtin_prefetch(Wo + (size_t)pn * DD + k + 32, 0, 1);
    }
    v16bf af[4];
#pragma unroll
    for (int mi = 0; mi < 4; ++mi) af[mi] = a_frag(At + mi * 16 * DD + k, lane);
#pragma unroll
    for (int i = 0; i < 4; ++i) {
      int n = ntb + i * 16 + nl;
      n = n < VDE ? n : VDE - 1;             // clamp OOB rows (edge block)
      v16bf b = *(const v16bf*)(Wo + (size_t)n * DD + k + hi * 16);
#pragma unroll
      for (int mi = 0; mi < 4; ++mi) acc[mi][i] = wmma_bf16(af[mi], b, acc[mi][i]);
    }
  }

#pragma unroll
  for (int mi = 0; mi < 4; ++mi) {
#pragma unroll
    for (int i = 0; i < 4; ++i) {
      int col = ntb + i * 16 + nl;
      if (col < VDE) {
#pragma unroll
        for (int r = 0; r < 8; ++r) {
          int row = m0 + mi * 16 + hi * 8 + r;
          out[(size_t)row * VDE + col] = acc[mi][i][r];
        }
      }
    }
  }
}

// ---------------------------------------------------------------------------
extern "C" void kernel_launch(void* const* d_in, const int* in_sizes, int n_in,
                              void* d_out, int out_size, void* d_ws,
                              size_t ws_size, hipStream_t stream) {
  const float* U    = (const float*)d_in[0];
  const float* b1   = (const float*)d_in[1];
  const float* b2   = (const float*)d_in[2];
  const float* bdec = (const float*)d_in[3];
  const float* Een  = (const float*)d_in[4];
  const float* Ede  = (const float*)d_in[5];
  const float* ctxW = (const float*)d_in[6];
  const float* Wout = (const float*)d_in[7];
  const int*   src  = (const int*)d_in[8];
  const int*   tgt  = (const int*)d_in[9];
  float* out = (float*)d_out;

  char* ws = (char*)d_ws;
  bf16*  Ut     = (bf16*)(ws + 0);              // 128 KB (transposed U)
  bf16*  Wc     = (bf16*)(ws + (1 << 17));      // 128 KB
  bf16*  Wo     = (bf16*)(ws + (1 << 18));      // 16,384,000 B
  float* H      = (float*)(ws + 16646144);      // 4 MB
  bf16*  outsBf = (bf16*)(ws + 20840448);       // 2 MB
  // total ~22.9 MB of workspace

  prep_kernel<<<8000, 256, 0, stream>>>(U, ctxW, Wout, Ut, Wc, Wo);
  encoder_kernel<<<1, 256, 0, stream>>>(Ut, Een, b1, b2, src, H);
  decoder_kernel<<<1, 256, 0, stream>>>(Ut, Wc, Ede, bdec, src, tgt, H, outsBf);
  logits_kernel<<<dim3(63, 64), 256, 0, stream>>>(outsBf, Wo, out);
}
